// GCET_44409961841160
// MI455X (gfx1250) — compile-verified, matrix operand order checked by reading, not run on hardware
//
#include <hip/hip_runtime.h>
#include <hip/hip_bf16.h>
#include <math.h>
#include <stdint.h>

// ---------------------------------------------------------------------------
// Types for CDNA5 WMMA (wave32): v_wmma_f32_16x16x32_bf16
// ---------------------------------------------------------------------------
typedef __attribute__((ext_vector_type(16))) __bf16 v16bf;
typedef __attribute__((ext_vector_type(8)))  float  v8f;

__device__ inline __bf16 f2bf(float f) { return static_cast<__bf16>(f); }

__device__ inline v8f wmma_bf16(v16bf a, v16bf b, v8f c) {
    return __builtin_amdgcn_wmma_f32_16x16x32_bf16(
        /*neg_a=*/false, a, /*neg_b=*/false, b,
        /*c_mod=*/(short)0, c, /*reuse_a=*/false, /*reuse_b=*/false);
}

// K index held by element e of a 16-bit A/B fragment, given lane-half.
// lanes 0-15: K = {0..7, 16..23}; lanes 16-31: K = {8..15, 24..31}
__device__ inline int kmap(int e, int half) {
    return ((e >> 3) << 4) + (half << 3) + (e & 7);
}

// Load a 16-element bf16 fragment from a k-contiguous fp32 row (stride 1 in k).
__device__ inline v16bf load_frag_contig(const float* __restrict__ row, int k0, int half) {
    const float4 a0 = *(const float4*)(row + k0 + half * 8);
    const float4 a1 = *(const float4*)(row + k0 + half * 8 + 4);
    const float4 a2 = *(const float4*)(row + k0 + 16 + half * 8);
    const float4 a3 = *(const float4*)(row + k0 + 16 + half * 8 + 4);
    v16bf f;
    f[0]  = f2bf(a0.x); f[1]  = f2bf(a0.y); f[2]  = f2bf(a0.z); f[3]  = f2bf(a0.w);
    f[4]  = f2bf(a1.x); f[5]  = f2bf(a1.y); f[6]  = f2bf(a1.z); f[7]  = f2bf(a1.w);
    f[8]  = f2bf(a2.x); f[9]  = f2bf(a2.y); f[10] = f2bf(a2.z); f[11] = f2bf(a2.w);
    f[12] = f2bf(a3.x); f[13] = f2bf(a3.y); f[14] = f2bf(a3.z); f[15] = f2bf(a3.w);
    return f;
}

// ---------------------------------------------------------------------------
// Generic batched WMMA GEMM (direct-load variant):
//   AM 0: A[m*K + k]            AM 1: A[(z*K + k)*M + m]     AM 2: A[(z*M+m)*K + k]
//   BM 0: B[(z*K + k)*P + p]    BM 1: B[(z*P + p)*K + k]
//   BM 2: conv (1,3) s3 im2col  BM 3: conv (1,3) over 3-wide rows
// Block = 128 threads (4 waves); wave -> one 16(M) x 64(P) tile.
// ---------------------------------------------------------------------------
template <int AM, int BM>
__global__ __launch_bounds__(128) void gemm_wmma_kernel(
    const float* __restrict__ A, const float* __restrict__ Bm,
    const float* __restrict__ bias, float* __restrict__ out,
    int M, int Kdim, int P, int bN)
{
    const int z    = blockIdx.z;
    const int wave = threadIdx.x >> 5;
    const int lane = threadIdx.x & 31;
    const int half = lane >> 4;
    const int lr   = lane & 15;
    const int m0   = (blockIdx.y * 4 + wave) * 16;
    if (m0 >= M) return;
    const long pbase = (long)blockIdx.x * 64;

    const int mrow  = m0 + lr;
    const int mload = mrow < M ? mrow : M - 1;

    long pl[4];
#pragma unroll
    for (int t = 0; t < 4; ++t) {
        long p = pbase + t * 16 + lr;
        pl[t] = p < P ? p : (long)P - 1;
    }

    const float* arow = nullptr;
    if (AM == 0) arow = A + (size_t)mload * Kdim;
    if (AM == 2) arow = A + ((size_t)z * M + mload) * Kdim;

    const float* brow[4];
    long boff[4];
    const int KC = Kdim / 3;
#pragma unroll
    for (int t = 0; t < 4; ++t) {
        if (BM == 0) brow[t] = Bm + (size_t)z * Kdim * P + pl[t];
        if (BM == 1) brow[t] = Bm + ((size_t)z * P + pl[t]) * Kdim;
        if (BM == 2) { long n = pl[t] / 3, w = pl[t] % 3;
                       brow[t] = Bm + (size_t)z * KC * bN * 9; boff[t] = n * 9 + w * 3; }
        if (BM == 3) { brow[t] = Bm + (size_t)z * KC * ((size_t)P * 3); boff[t] = pl[t] * 3; }
    }

    v8f acc[4] = {};

    for (int k0 = 0; k0 < Kdim; k0 += 32) {
        v16bf af;
        if (AM == 0 || AM == 2) {
            af = load_frag_contig(arow, k0, half);
        } else { // AM == 1: strided by M
#pragma unroll
            for (int e = 0; e < 16; ++e) {
                int k = k0 + kmap(e, half);
                af[e] = f2bf(A[((size_t)z * Kdim + k) * M + mload]);
            }
        }
#pragma unroll
        for (int t = 0; t < 4; ++t) {
            v16bf bf;
            if (BM == 1) {
                bf = load_frag_contig(brow[t], k0, half);
            } else {
#pragma unroll
                for (int e = 0; e < 16; ++e) {
                    int k = k0 + kmap(e, half);
                    float v;
                    if (BM == 0) v = brow[t][(size_t)k * P];
                    else if (BM == 2) { int c = k / 3, tt = k % 3;
                                        v = brow[t][(size_t)c * bN * 9 + boff[t] + tt]; }
                    else              { int c = k / 3, tt = k % 3;
                                        v = brow[t][(size_t)c * P * 3 + boff[t] + tt]; }
                    bf[e] = f2bf(v);
                }
            }
            acc[t] = wmma_bf16(af, bf, acc[t]);
        }
    }
#pragma unroll
    for (int t = 0; t < 4; ++t) {
        long p = pbase + t * 16 + lr;
        if (p >= P) continue;
#pragma unroll
        for (int v = 0; v < 8; ++v) {
            int m = m0 + v + half * 8;
            if (m < M) {
                float r = acc[t][v] + (bias ? bias[m] : 0.0f);
                out[((size_t)z * M + m) * P + p] = r;
            }
        }
    }
}

// ---------------------------------------------------------------------------
// LDS-staged BM0 GEMM: B[(z*K+k)*P + p]. The 32(K)x64(P) B tile is staged
// into LDS once per block per k-step with CDNA5 async-to-LDS loads
// (global_load_async_to_lds_b128, ASYNCcnt), then all 4 waves read their
// fragments from LDS. Requires P >= 64-ish (host dispatches only when P>=64).
// All waves stay resident for the barrier pipeline; invalid m-tiles clamp
// loads and mask stores (EXEC stays all-ones for WMMA).
// ---------------------------------------------------------------------------
template <int AM>
__global__ __launch_bounds__(128) void gemm_wmma_smem_kernel(
    const float* __restrict__ A, const float* __restrict__ Bm,
    const float* __restrict__ bias, float* __restrict__ out,
    int M, int Kdim, int P)
{
    __shared__ float btile[32 * 64];

    const int z    = blockIdx.z;
    const int wave = threadIdx.x >> 5;
    const int lane = threadIdx.x & 31;
    const int half = lane >> 4;
    const int lr   = lane & 15;
    const int m0   = (blockIdx.y * 4 + wave) * 16;
    const long pbase = (long)blockIdx.x * 64;

    const int mrow  = m0 + lr;
    const int mclmp = mrow < M ? mrow : M - 1;

    const float* arow = nullptr;
    if (AM == 0) arow = A + (size_t)mclmp * Kdim;
    if (AM == 2) arow = A + ((size_t)z * M + mclmp) * Kdim;

    const float* bz = Bm + (size_t)z * Kdim * P;
    long pmax = (long)P - 4; if (pmax < 0) pmax = 0;

    // per-thread staging geometry: 4 float4s, covering 32 rows x 64 cols
    int f0   = threadIdx.x;            // float4 id base; f = l*128 + f0
    v8f acc[4] = {};

    for (int k0 = 0; k0 < Kdim; k0 += 32) {
        // ---- stage B tile (async global -> LDS) ----
#pragma unroll
        for (int l = 0; l < 4; ++l) {
            int fidx = l * 128 + f0;               // 0..511
            int kk   = fidx >> 4;                  // 0..31
            int c4   = (fidx & 15) * 4;            // 0..60
            long pc  = pbase + c4;
            if (pc > pmax) pc = pmax;
            const float* src = bz + (size_t)(k0 + kk) * P + pc;
            unsigned dst = (unsigned)(uintptr_t)(&btile[kk * 64 + c4]);
            unsigned long long ga = (unsigned long long)(uintptr_t)src;
            asm volatile("global_load_async_to_lds_b128 %0, %1, off"
                         :: "v"(dst), "v"(ga) : "memory");
        }
        asm volatile("s_wait_asynccnt 0x0" ::: "memory");
        __syncthreads();

        // ---- A fragment ----
        v16bf af;
        if (AM == 0 || AM == 2) {
            af = load_frag_contig(arow, k0, half);
        } else { // AM == 1: strided by M
#pragma unroll
            for (int e = 0; e < 16; ++e) {
                int k = k0 + kmap(e, half);
                af[e] = f2bf(A[((size_t)z * Kdim + k) * M + mclmp]);
            }
        }
        // ---- B fragments from LDS + WMMA ----
#pragma unroll
        for (int t = 0; t < 4; ++t) {
            v16bf bf;
#pragma unroll
            for (int e = 0; e < 16; ++e) {
                int k = kmap(e, half);
                bf[e] = f2bf(btile[k * 64 + t * 16 + lr]);
            }
            acc[t] = wmma_bf16(af, bf, acc[t]);
        }
        __syncthreads();   // protect tile before next overwrite
    }
#pragma unroll
    for (int t = 0; t < 4; ++t) {
        long p = pbase + t * 16 + lr;
        if (p >= P) continue;
#pragma unroll
        for (int v = 0; v < 8; ++v) {
            int m = m0 + v + half * 8;
            if (m < M) {
                float r = acc[t][v] + (bias ? bias[m] : 0.0f);
                out[((size_t)z * M + m) * P + p] = r;
            }
        }
    }
}

// ---------------------------------------------------------------------------
// xx[b,n] = sum_c x[b,c,n]^2
// ---------------------------------------------------------------------------
__global__ void sumsq_kernel(const float* __restrict__ x, float* __restrict__ xx, int N)
{
    int z = blockIdx.y;
    int n = blockIdx.x * blockDim.x + threadIdx.x;
    if (n >= N) return;
    float s = 0.f;
    for (int c = 0; c < 128; ++c) {
        float v = x[((size_t)z * 128 + c) * N + n];
        s += v * v;
    }
    xx[(size_t)z * N + n] = s;
}

// ---------------------------------------------------------------------------
// Fused Gram + top-9 KNN. pd(i,j) = 2*x_i.x_j - |x_i|^2 - |x_j|^2.
// ---------------------------------------------------------------------------
__global__ __launch_bounds__(256) void knn_topk_kernel(
    const float* __restrict__ x, const float* __restrict__ xx,
    int* __restrict__ idx, int N)
{
    const int C = 128;
    __shared__ float sc[8][16][17];
    __shared__ float msv[8][16][9];
    __shared__ int   msi[8][16][9];

    const int z    = blockIdx.y;
    const int wave = threadIdx.x >> 5;
    const int lane = threadIdx.x & 31;
    const int half = lane >> 4;
    const int lr   = lane & 15;
    const int m0   = blockIdx.x * 128 + wave * 16;
    if (m0 >= N) return;

    const int row   = m0 + lr;
    const int rload = row < N ? row : N - 1;

    v16bf af[4];
#pragma unroll
    for (int kc = 0; kc < 4; ++kc)
#pragma unroll
        for (int e = 0; e < 16; ++e) {
            int c = kc * 32 + kmap(e, half);
            af[kc][e] = f2bf(x[((size_t)z * C + c) * N + rload]);
        }
    float xxr[8];
#pragma unroll
    for (int v = 0; v < 8; ++v) {
        int m = m0 + v + half * 8;
        xxr[v] = xx[(size_t)z * N + (m < N ? m : N - 1)];
    }
    float ts[9]; int ti[9];
#pragma unroll
    for (int q = 0; q < 9; ++q) { ts[q] = -3.4e38f; ti[q] = 0x7fffffff; }

    for (int j0 = 0; j0 < N; j0 += 16) {
        int pc = j0 + lr;
        int pl = pc < N ? pc : N - 1;
        v8f acc = {};
#pragma unroll
        for (int kc = 0; kc < 4; ++kc) {
            v16bf bf;
#pragma unroll
            for (int e = 0; e < 16; ++e) {
                int c = kc * 32 + kmap(e, half);
                bf[e] = f2bf(x[((size_t)z * C + c) * N + pl]);
            }
            acc = wmma_bf16(af[kc], bf, acc);
        }
        float xxj = xx[(size_t)z * N + pl];
#pragma unroll
        for (int v = 0; v < 8; ++v)
            sc[wave][v + half * 8][lr] = 2.0f * acc[v] - xxr[v] - xxj;
        __builtin_amdgcn_wave_barrier();
        __threadfence_block();
#pragma unroll
        for (int cc = 0; cc < 8; ++cc) {
            int j = j0 + half * 8 + cc;
            if (j < N) {
                float s = sc[wave][lr][half * 8 + cc];
                if (s > ts[8] || (s == ts[8] && j < ti[8])) {
                    ts[8] = s; ti[8] = j;
#pragma unroll
                    for (int q = 8; q > 0; --q) {
                        if (ts[q] > ts[q - 1] || (ts[q] == ts[q - 1] && ti[q] < ti[q - 1])) {
                            float tv = ts[q]; ts[q] = ts[q - 1]; ts[q - 1] = tv;
                            int tj = ti[q]; ti[q] = ti[q - 1]; ti[q - 1] = tj;
                        } else break;
                    }
                }
            }
        }
        __builtin_amdgcn_wave_barrier();
        __threadfence_block();
    }
    if (half == 1) {
#pragma unroll
        for (int q = 0; q < 9; ++q) { msv[wave][lr][q] = ts[q]; msi[wave][lr][q] = ti[q]; }
    }
    __builtin_amdgcn_wave_barrier();
    __threadfence_block();
    if (half == 0 && row < N) {
        int ia = 0, ib = 0;
#pragma unroll
        for (int q = 0; q < 9; ++q) {               // ia+ib == q <= 8, no OOB
            float sb = msv[wave][lr][ib]; int jb = msi[wave][lr][ib];
            bool takeA = (ts[ia] > sb || (ts[ia] == sb && ti[ia] < jb));
            int oj;
            if (takeA) { oj = ti[ia]; ++ia; } else { oj = jb; ++ib; }
            idx[((size_t)z * N + row) * 9 + q] = oj;
        }
    }
}

// ---------------------------------------------------------------------------
// g[b, c2, n, k]:  c2<128 -> x[b,c2,n];  c2>=128 -> x[b,c,n]-x[b,c,idx[b,n,k]]
// ---------------------------------------------------------------------------
__global__ void build_graph_kernel(const float* __restrict__ x, const int* __restrict__ idx,
                                   float* __restrict__ g, int N)
{
    int z = blockIdx.z, c2 = blockIdx.y;
    long pos = (long)blockIdx.x * blockDim.x + threadIdx.x;
    if (pos >= (long)N * 9) return;
    int n = (int)(pos / 9), kk = (int)(pos % 9);
    float val;
    if (c2 < 128) {
        val = x[((size_t)z * 128 + c2) * N + n];
    } else {
        int c = c2 - 128;
        int j = idx[((size_t)z * N + n) * 9 + kk];
        val = x[((size_t)z * 128 + c) * N + n] - x[((size_t)z * 128 + c) * N + j];
    }
    g[(((size_t)z * 256 + c2) * N + n) * 9 + kk] = val;
}

// ---------------------------------------------------------------------------
// mean/biased-variance per (b,c) (per_bc=1) or per c over batch (per_bc=0)
// ---------------------------------------------------------------------------
__global__ __launch_bounds__(256) void stats_kernel(
    const float* __restrict__ x, float* __restrict__ mean, float* __restrict__ var,
    int C, long P, int Bn, int per_bc)
{
    __shared__ float ss[256], sq[256];
    int c = blockIdx.x;
    float s = 0.f, q = 0.f;
    if (per_bc) {
        int b = blockIdx.y;
        const float* base = x + ((size_t)b * C + c) * P;
        for (long i = threadIdx.x; i < P; i += 256) { float v = base[i]; s += v; q += v * v; }
    } else {
        for (int b = 0; b < Bn; ++b) {
            const float* base = x + ((size_t)b * C + c) * P;
            for (long i = threadIdx.x; i < P; i += 256) { float v = base[i]; s += v; q += v * v; }
        }
    }
    ss[threadIdx.x] = s; sq[threadIdx.x] = q;
    __syncthreads();
    for (int o = 128; o > 0; o >>= 1) {
        if (threadIdx.x < o) { ss[threadIdx.x] += ss[threadIdx.x + o]; sq[threadIdx.x] += sq[threadIdx.x + o]; }
        __syncthreads();
    }
    if (threadIdx.x == 0) {
        float cnt = per_bc ? (float)P : (float)P * Bn;
        float m = ss[0] / cnt;
        int gidx = per_bc ? blockIdx.y * C + c : c;
        mean[gidx] = m;
        var[gidx]  = sq[0] / cnt - m * m;
    }
}

// IN(eps_in) followed by BN(1e-5): combined per-(b,c) scale.
__global__ void finalize_inbn_kernel(const float* __restrict__ var, float* __restrict__ scale,
                                     int C, int Bn, float eps_in)
{
    int c = blockIdx.x * blockDim.x + threadIdx.x;
    if (c >= C) return;
    float vb = 0.f;
    for (int b = 0; b < Bn; ++b) { float v = var[b * C + c]; vb += v / (v + eps_in); }
    vb /= Bn;
    float rs = rsqrtf(vb + 1e-5f);
    for (int b = 0; b < Bn; ++b) scale[b * C + c] = rsqrtf(var[b * C + c] + eps_in) * rs;
}

__global__ void finalize_bn_kernel(const float* __restrict__ var, float* __restrict__ scale,
                                   int C, float eps)
{
    int c = blockIdx.x * blockDim.x + threadIdx.x;
    if (c >= C) return;
    scale[c] = rsqrtf(var[c] + eps);
}

__global__ void apply_norm_kernel(const float* __restrict__ x, float* __restrict__ out,
                                  const float* __restrict__ mean, const float* __restrict__ scale,
                                  int C, long P, int per_bc, int relu)
{
    int c = blockIdx.y, b = blockIdx.z;
    long i = (long)blockIdx.x * blockDim.x + threadIdx.x;
    if (i >= P) return;
    int g = per_bc ? b * C + c : c;
    float v = (x[((size_t)b * C + c) * P + i] - mean[g]) * scale[g];
    if (relu) v = v > 0.f ? v : 0.f;
    out[((size_t)b * C + c) * P + i] = v;
}

// ---------------------------------------------------------------------------
// Elementwise helpers
// ---------------------------------------------------------------------------
__global__ void add_kernel(const float* __restrict__ a, const float* __restrict__ b,
                           float* __restrict__ o, long n, int relu)
{
    long i = (long)blockIdx.x * blockDim.x + threadIdx.x;
    if (i >= n) return;
    float v = a[i] + b[i];
    if (relu) v = fmaxf(v, 0.f);
    o[i] = v;
}

__global__ void axpy_gamma_kernel(const float* __restrict__ base, const float* __restrict__ o,
                                  const float* __restrict__ gamma, float* __restrict__ out, long n)
{
    long i = (long)blockIdx.x * blockDim.x + threadIdx.x;
    if (i >= n) return;
    out[i] = base[i] + gamma[0] * o[i];
}

__global__ void aff_combine_kernel(const float* __restrict__ xv, const float* __restrict__ rv,
                                   const float* __restrict__ xl, const float* __restrict__ xg,
                                   float* __restrict__ out, int C, long P)
{
    int c = blockIdx.y, b = blockIdx.z;
    long i = (long)blockIdx.x * blockDim.x + threadIdx.x;
    if (i >= P) return;
    size_t id = ((size_t)b * C + c) * P + i;
    float w = xl[id] + xg[b * C + c];
    w = 1.0f / (1.0f + __expf(-w));
    out[id] = 2.0f * xv[id] * w + 2.0f * rv[id] * (1.0f - w);
}

__global__ void max9_kernel(const float* __restrict__ in, float* __restrict__ out, int Cch, int N)
{
    int c = blockIdx.y, b = blockIdx.z;
    int n = blockIdx.x * blockDim.x + threadIdx.x;
    if (n >= N) return;
    const float* p = in + (((size_t)b * Cch + c) * N + n) * 9;
    float m = p[0];
#pragma unroll
    for (int k = 1; k < 9; ++k) m = fmaxf(m, p[k]);
    out[((size_t)b * Cch + c) * N + n] = m;
}

// softmax along a contiguous row of length L (one block per row, in place)
__global__ __launch_bounds__(256) void softmax_row_kernel(float* __restrict__ x, long L)
{
    __shared__ float red[256];
    size_t base = (size_t)blockIdx.x * L;
    float m = -3.4e38f;
    for (long i = threadIdx.x; i < L; i += 256) m = fmaxf(m, x[base + i]);
    red[threadIdx.x] = m; __syncthreads();
    for (int o = 128; o > 0; o >>= 1) {
        if (threadIdx.x < o) red[threadIdx.x] = fmaxf(red[threadIdx.x], red[threadIdx.x + o]);
        __syncthreads();
    }
    m = red[0]; __syncthreads();
    float s = 0.f;
    for (long i = threadIdx.x; i < L; i += 256) {
        float e = __expf(x[base + i] - m); x[base + i] = e; s += e;
    }
    red[threadIdx.x] = s; __syncthreads();
    for (int o = 128; o > 0; o >>= 1) {
        if (threadIdx.x < o) red[threadIdx.x] += red[threadIdx.x + o];
        __syncthreads();
    }
    float inv = 1.0f / red[0];
    for (long i = threadIdx.x; i < L; i += 256) x[base + i] *= inv;
}

// softmax over the channel dim (stride N apart), one block per (b,n), Lc<=256
__global__ __launch_bounds__(256) void softmax_col_kernel(float* __restrict__ x, int Lc, int N)
{
    __shared__ float red[256];
    int n = blockIdx.x, b = blockIdx.y, l = threadIdx.x;
    float v = (l < Lc) ? x[((size_t)b * Lc + l) * N + n] : -3.4e38f;
    red[l] = v; __syncthreads();
    for (int o = 128; o > 0; o >>= 1) {
        if (l < o) red[l] = fmaxf(red[l], red[l + o]);
        __syncthreads();
    }
    float m = red[0]; __syncthreads();
    float e = (l < Lc) ? __expf(v - m) : 0.f;
    red[l] = e; __syncthreads();
    for (int o = 128; o > 0; o >>= 1) {
        if (l < o) red[l] += red[l + o];
        __syncthreads();
    }
    float inv = 1.0f / red[0];
    if (l < Lc) x[((size_t)b * Lc + l) * N + n] = e * inv;
}

// ---------------------------------------------------------------------------
// Host orchestration
// ---------------------------------------------------------------------------
extern "C" void kernel_launch(void* const* d_in, const int* in_sizes, int n_in,
                              void* d_out, int out_size, void* d_ws, size_t ws_size,
                              hipStream_t stream)
{
    (void)in_sizes; (void)n_in; (void)out_size; (void)ws_size;
    const int N = 4000, Bn = 2;
    auto PF = [&](int i) { return (const float*)d_in[i]; };
    const float* X = PF(0);   // (B,128,N)

    // -------- workspace carving (floats) --------
    float* W = (float*)d_ws;
    size_t off = 0;
    auto alloc = [&](size_t nf) { float* p = W + off; off += nf; return p; };

    float* XX   = alloc(8000);
    int*   IDX  = (int*)alloc(72000);
    float* G    = alloc(18432000);          // (B,256,N,9)
    float* BIG1 = alloc(18432000);
    float* BIG2 = alloc(18432000);
    float* AN   = alloc(1024000);           // (B,128,N)
    float* MXP  = alloc(2048000);           // (B,256,N)
    float* MX   = alloc(1024000);
    float* SAb  = alloc(2048000);
    float* SBb  = alloc(2048000);
    float* SCb  = alloc(2048000);
    float* S1   = alloc(2048000);
    float* S2   = alloc(2048000);
    float* S3   = alloc(2048000);
    float* S4   = alloc(2048000);
    float* CLA  = alloc(65536);
    float* CLM  = alloc(65536);
    float* I1   = alloc(65536);
    float* I2   = alloc(65536);
    float* I3   = alloc(65536);
    float* I4   = alloc(65536);
    float* QB   = alloc(16384);
    float* KB   = alloc(16384);
    float* VB   = alloc(65536);
    float* OB   = alloc(65536);
    float* O2   = alloc(65536);
    float* ATT  = alloc(131072);
    float* XAs  = alloc(65536);
    float* XLs  = alloc(65536);
    float* L1s  = alloc(16384);
    float* G1B  = alloc(64);
    float* XGB  = alloc(256);
    float* F1   = alloc(65536);
    float* F2   = alloc(65536);
    float* OMX  = alloc(1024000);
    float* OAN  = alloc(1024000);
    float* MEAN = alloc(512);
    float* VAR  = alloc(512);
    float* SCL  = alloc(512);
    float* GM   = alloc(256);
    float* VARD = alloc(512);

    // -------- helpers --------
    auto gemm = [&](const float* A, const float* Bm, const float* bias, float* out,
                    int M, int K, int P, int amode, int bmode, int bN) {
        dim3 grid((P + 63) / 64, (M + 63) / 64, Bn);
        if (bmode == 0 && P >= 64) {              // LDS-staged async path
            if (amode == 0)
                gemm_wmma_smem_kernel<0><<<grid, 128, 0, stream>>>(A, Bm, bias, out, M, K, P);
            else if (amode == 1)
                gemm_wmma_smem_kernel<1><<<grid, 128, 0, stream>>>(A, Bm, bias, out, M, K, P);
            else
                gemm_wmma_smem_kernel<2><<<grid, 128, 0, stream>>>(A, Bm, bias, out, M, K, P);
        } else if (amode == 0 && bmode == 0) {
            gemm_wmma_kernel<0, 0><<<grid, 128, 0, stream>>>(A, Bm, bias, out, M, K, P, bN);
        } else if (amode == 0 && bmode == 2) {
            gemm_wmma_kernel<0, 2><<<grid, 128, 0, stream>>>(A, Bm, bias, out, M, K, P, bN);
        } else if (amode == 0 && bmode == 3) {
            gemm_wmma_kernel<0, 3><<<grid, 128, 0, stream>>>(A, Bm, bias, out, M, K, P, bN);
        } else { // amode == 2 && bmode == 1
            gemm_wmma_kernel<2, 1><<<grid, 128, 0, stream>>>(A, Bm, bias, out, M, K, P, bN);
        }
    };
    auto bn_apply = [&](float* xbuf, int Cch, long P, int relu) {
        stats_kernel<<<dim3(Cch, 1), 256, 0, stream>>>(xbuf, MEAN, VAR, Cch, P, Bn, 0);
        finalize_bn_kernel<<<dim3((Cch + 63) / 64), 64, 0, stream>>>(VAR, SCL, Cch, 1e-5f);
        apply_norm_kernel<<<dim3((unsigned)((P + 255) / 256), Cch, Bn), 256, 0, stream>>>(
            xbuf, xbuf, MEAN, SCL, Cch, P, 0, relu);
    };
    auto inbn_apply = [&](const float* xin, float* xout, int Cch, long P, float eps_in, int relu) {
        stats_kernel<<<dim3(Cch, Bn), 256, 0, stream>>>(xin, MEAN, VAR, Cch, P, Bn, 1);
        finalize_inbn_kernel<<<dim3((Cch + 63) / 64), 64, 0, stream>>>(VAR, SCL, Cch, Bn, eps_in);
        apply_norm_kernel<<<dim3((unsigned)((P + 255) / 256), Cch, Bn), 256, 0, stream>>>(
            xin, xout, MEAN, SCL, Cch, P, 1, relu);
    };
    auto ew_add = [&](const float* a, const float* b, float* o, long n, int relu) {
        add_kernel<<<dim3((unsigned)((n + 255) / 256)), 256, 0, stream>>>(a, b, o, n, relu);
    };
    auto res_block = [&](const float* Xin, int ic, int oc, long P,
                         const float* rw, const float* rb, const float* w1, const float* b1,
                         const float* w2, const float* b2,
                         float* T1, float* T2, float* T3, float* out) {
        gemm(rw, Xin, rb, T1, oc, ic, (int)P, 0, 0, 0);
        gemm(w1, Xin, b1, T2, oc, ic, (int)P, 0, 0, 0);
        inbn_apply(T2, T2, oc, P, 1e-5f, 1);
        gemm(w2, T2, b2, T3, oc, oc, (int)P, 0, 0, 0);
        inbn_apply(T3, T3, oc, P, 1e-5f, 0);
        ew_add(T3, T1, out, (long)Bn * oc * P, 1);
    };
    auto inter = [&](const float* n1, const float* n2, const float* n3, int pi, float* out) {
        const float* cb = PF(pi + 0); const float* cw = PF(pi + 1); const float* gm = PF(pi + 2);
        const float* kb = PF(pi + 3); const float* kw = PF(pi + 4);
        const float* qb = PF(pi + 5); const float* qw = PF(pi + 6);
        const float* vb = PF(pi + 7); const float* vw = PF(pi + 8);
        gemm(qw, n1, qb, QB, 32, 128, 256, 0, 0, 0);  bn_apply(QB, 32, 256, 1);
        gemm(kw, n2, kb, KB, 32, 128, 256, 0, 0, 0);  bn_apply(KB, 32, 256, 1);
        gemm(vw, n3, vb, VB, 128, 128, 256, 0, 0, 0); bn_apply(VB, 128, 256, 1);
        gemm(QB, KB, nullptr, ATT, 256, 32, 256, 1, 0, 0);          // att = q^T k
        softmax_row_kernel<<<dim3(Bn * 256), 256, 0, stream>>>(ATT, 256);
        gemm(VB, ATT, nullptr, OB, 128, 256, 256, 2, 1, 0);         // o = v att^T
        gemm(cw, OB, cb, O2, 128, 128, 256, 0, 0, 0);  bn_apply(O2, 128, 256, 1);
        axpy_gamma_kernel<<<dim3((65536 + 255) / 256), 256, 0, stream>>>(n3, O2, gm, out, 65536);
    };
    auto aff = [&](const float* xv, const float* rv, int pi, long P,
                   float* xa, float* xl, float* l1, float* out) {
        const float* gb1 = PF(pi + 0); const float* gb2 = PF(pi + 1);
        const float* gw1 = PF(pi + 2); const float* gw2 = PF(pi + 3);
        const float* lb1 = PF(pi + 4); const float* lb2 = PF(pi + 5);
        const float* lw1 = PF(pi + 6); const float* lw2 = PF(pi + 7);
        ew_add(xv, rv, xa, (long)Bn * 128 * P, 0);
        stats_kernel<<<dim3(128, Bn), 256, 0, stream>>>(xa, GM, VARD, 128, P, Bn, 1); // GM = avg pool
        gemm(lw1, xa, lb1, l1, 32, 128, (int)P, 0, 0, 0);  bn_apply(l1, 32, P, 1);
        gemm(lw2, l1, lb2, xl, 128, 32, (int)P, 0, 0, 0);  bn_apply(xl, 128, P, 0);
        gemm(gw1, GM, gb1, G1B, 32, 128, 1, 0, 0, 0);      bn_apply(G1B, 32, 1, 1);
        gemm(gw2, G1B, gb2, XGB, 128, 32, 1, 0, 0, 0);     bn_apply(XGB, 128, 1, 0);
        aff_combine_kernel<<<dim3((unsigned)((P + 255) / 256), 128, Bn), 256, 0, stream>>>(
            xv, rv, xl, XGB, out, 128, P);
    };
    auto group = [&](const float* Xin, int pib, int piw, float* Sout, float* tmp) {
        inbn_apply(Xin, tmp, 128, N, 1e-3f, 1);
        gemm(PF(piw), tmp, PF(pib), Sout, 256, 128, N, 0, 0, 0);
    };

    // ======================= pipeline =======================
    // 1. KNN graph
    sumsq_kernel<<<dim3((N + 255) / 256, Bn), 256, 0, stream>>>(X, XX, N);
    knn_topk_kernel<<<dim3((N + 127) / 128, Bn), 256, 0, stream>>>(X, XX, IDX, N);
    build_graph_kernel<<<dim3((N * 9 + 255) / 256, 256, Bn), 256, 0, stream>>>(X, IDX, G, N);

    // 2. convseq: (1,3)s3 conv -> BN+ReLU -> (1,3) conv -> BN+ReLU
    float* CS1 = BIG1;   // (B,256,12000)
    float* CS2 = BIG2;   // (B,256,4000)
    gemm(PF(39), G, PF(37), CS1, 256, 768, 12000, 0, 2, N);
    bn_apply(CS1, 256, 12000, 1);
    gemm(PF(40), CS1, PF(38), CS2, 256, 768, 4000, 0, 3, 0);
    bn_apply(CS2, 256, 4000, 1);

    // 3. out_an = Res(change1)(convseq)
    res_block(CS2, 256, 128, N, PF(28), PF(27), PF(29), PF(25), PF(30), PF(26),
              SAb, SBb, SCb, AN);

    // 4. Res(mlp1)(g) then max over k
    gemm(PF(88), G, PF(87), BIG1, 256, 256, 36000, 0, 0, 0);   // x1
    gemm(PF(89), G, PF(85), BIG2, 256, 256, 36000, 0, 0, 0);   // h
    inbn_apply(BIG2, BIG2, 256, 36000, 1e-5f, 1);
    gemm(PF(90), BIG2, PF(86), G, 256, 256, 36000, 0, 0, 0);   // reuse G
    inbn_apply(G, G, 256, 36000, 1e-5f, 0);
    ew_add(G, BIG1, BIG2, (long)Bn * 256 * 36000, 1);
    max9_kernel<<<dim3((N + 255) / 256, 256, Bn), 256, 0, stream>>>(BIG2, MXP, 256, N);

    // 5. out_max = Res(change2)(max)
    res_block(MXP, 256, 128, N, PF(34), PF(33), PF(35), PF(31), PF(36), PF(32),
              SAb, SBb, SCb, MX);

    // 6. grouping softmaxes
    group(AN, 41, 42, S1, SAb);
    softmax_row_kernel<<<dim3(Bn * 256), 256, 0, stream>>>(S1, N);
    group(MX, 43, 44, S2, SAb);
    softmax_row_kernel<<<dim3(Bn * 256), 256, 0, stream>>>(S2, N);
    group(MX, 45, 46, S3, SAb);
    softmax_col_kernel<<<dim3(N, Bn), 256, 0, stream>>>(S3, 256, N);
    group(AN, 47, 48, S4, SAb);
    softmax_col_kernel<<<dim3(N, Bn), 256, 0, stream>>>(S4, 256, N);

    // 7. cluster pooling: bcn,bln->bcl
    gemm(AN, S1, nullptr, CLA, 128, 4000, 256, 2, 1, 0);
    gemm(MX, S2, nullptr, CLM, 128, 4000, 256, 2, 1, 0);

    // 8. attention blocks on clusters
    inter(CLA, CLA, CLA, 49, I1);   // inter1
    inter(CLM, CLM, CLM, 58, I2);   // inter2
    inter(CLA, CLM, CLM, 67, I3);   // intra1
    inter(CLM, CLA, CLA, 76, I4);   // intra2

    // 9. AFF fusions on clusters
    aff(I1, I3, 1, 256, XAs, XLs, L1s, F1);   // aff1
    aff(I2, I4, 9, 256, XAs, XLs, L1s, F2);   // aff2

    // 10. unpool: bcl,bln->bcn
    gemm(F1, S3, nullptr, OMX, 128, 256, 4000, 2, 0, 0);
    gemm(F2, S4, nullptr, OAN, 128, 256, 4000, 2, 0, 0);

    // 11. final AFF -> d_out (B,128,N,1)
    aff(OMX, OAN, 17, N, SAb, SBb, SCb, (float*)d_out);
}
